// ClockHConvGRUCell_79860621902135
// MI455X (gfx1250) — matrix operand order repaired and checked by reading, not compiled, not run on hardware
//
#include <hip/hip_runtime.h>
#include <hip/hip_bf16.h>
#include <math.h>

#define HCH 128               // hidden channels
#define SS  96                // spatial
#define BB  8                 // batch
#define HW  (SS*SS)           // 9216
#define NPIX (BB*HW)          // 73728
#define NTOT ((size_t)HCH*NPIX)
#define PAD 2
#define EPSB 1e-3f

typedef __attribute__((ext_vector_type(16))) __bf16 v16bf;
typedef __attribute__((ext_vector_type(8)))  float  v8f;

union BF16x16 { v16bf v; unsigned int u[8]; uint4 q[2]; };

__device__ __forceinline__ unsigned short f2bf(float f) {
  unsigned int u = __float_as_uint(f);
  unsigned int r = u + 0x7FFFu + ((u >> 16) & 1u);   // round-to-nearest-even
  return (unsigned short)(r >> 16);
}
__device__ __forceinline__ unsigned int pack2bf(float lo, float hi) {
  return (unsigned int)f2bf(lo) | ((unsigned int)f2bf(hi) << 16);
}
__device__ __forceinline__ float sigmoidf_(float x) { return 1.0f / (1.0f + __expf(-x)); }
__device__ __forceinline__ int frag_k(int hi, int j) {      // A 16x32 bf16 K index (ISA 7.12.2)
  return hi * 8 + ((j < 4) ? 2 * j : 16 + 2 * (j - 4));
}

// ---------------------------------------------------------------------------
// Pack a 128x128 f32 matrix into WMMA A-fragment image:
//   out[((chunk*8 + wave)*32 + lane)*8 + j] = bf16 pair (K=kidx, kidx+1)
// ---------------------------------------------------------------------------
__global__ void pack_w1_kernel(const float* __restrict__ src, unsigned int* __restrict__ dst) {
  int tid = blockIdx.x * 256 + threadIdx.x;              // 8192
  if (tid >= 8192) return;
  int j = tid & 7, lane = (tid >> 3) & 31, wave = (tid >> 8) & 7, chunk = tid >> 11;
  int row = lane & 15, hi = lane >> 4;
  int o = wave * 16 + row;
  int k = chunk * 32 + frag_k(hi, j);
  dst[tid] = pack2bf(src[(size_t)o * HCH + k], src[(size_t)o * HCH + k + 1]);
}

// ---------------------------------------------------------------------------
// Pack 5x5 conv weights [o][i][5][5] into per-tap A-fragment image:
//   out[(((t*4 + chunk)*8 + wave)*32 + lane)*8 + j]   (25*8192 u32 = 800 KB)
// ---------------------------------------------------------------------------
__global__ void pack_wk_kernel(const float* __restrict__ src, unsigned int* __restrict__ dst) {
  int tid = blockIdx.x * 256 + threadIdx.x;              // 204800 = 25 * 2^13
  if (tid >= 25 * 8192) return;
  int j = tid & 7, lane = (tid >> 3) & 31, wave = (tid >> 8) & 7;
  int chunk = (tid >> 11) & 3, t = tid >> 13;
  int row = lane & 15, hi = lane >> 4;
  int o = wave * 16 + row;
  int i = chunk * 32 + frag_k(hi, j);
  dst[tid] = pack2bf(src[((size_t)o * HCH + i) * 25 + t],
                     src[((size_t)o * HCH + i + 1) * 25 + t]);
}

// ---------------------------------------------------------------------------
// Fused 1x1-conv (3 GEMMs) + gate epilogue. Two pixel-tiles per wave so each
// A fragment feeds two WMMAs (halves weight traffic).
// MODE 0: att gate   -> gbuf = e * gate (bf16)
// MODE 1: inh update -> d_out[0] = inh_new (f32), gbuf bf16
// MODE 2: exc update -> d_out[1] = exc_new (f32)
// Block: 256 thr = 8 waves; wave w: out-channels [16w,16w+16) x 32 pixels.
// ---------------------------------------------------------------------------
template<int MODE>
__global__ __launch_bounds__(256)
void gate_kernel(const float* __restrict__ srcB0, const float* __restrict__ srcB1,
                 const unsigned int* __restrict__ wA0,
                 const unsigned int* __restrict__ wA1,
                 const unsigned int* __restrict__ wA2,
                 const float* __restrict__ bias0, const float* __restrict__ bias1,
                 const float* __restrict__ bias2,
                 const int* __restrict__ pstep,
                 const float* __restrict__ convraw, const float* __restrict__ stats,
                 const float* __restrict__ bnw, const float* __restrict__ bnb,
                 const float* __restrict__ c0, const float* __restrict__ c1,
                 const float* __restrict__ c2,
                 float* __restrict__ outF, unsigned short* __restrict__ outBF) {
  __shared__ unsigned short sB0[32 * HCH];   // [col][k]  (K-pairs contiguous)
  __shared__ unsigned short sB1[32 * HCH];

  const int tid  = threadIdx.x;
  const int p0   = blockIdx.x * 32;          // flat pixel = b*HW + hw
  const int b    = p0 / HW;
  const int hw0  = p0 - b * HW;
  const size_t gbase = (size_t)b * HCH * HW + hw0;

  for (int idx = tid; idx < 32 * HCH; idx += 256) {
    int c = idx >> 5, j = idx & 31;
    sB0[j * HCH + c] = f2bf(srcB0[gbase + (size_t)c * HW + j]);
    sB1[j * HCH + c] = f2bf(srcB1[gbase + (size_t)c * HW + j]);
  }
  __syncthreads();

  const int wave = tid >> 5, lane = tid & 31;
  const int m0 = wave * 16;
  const int row = lane & 15, hi = lane >> 4;
  const size_t lbase = ((size_t)wave * 32 + lane) * 8;

  v8f accW[2] = {{}, {}}, accU[2] = {{}, {}}, accC[2] = {{}, {}};
  for (int kk = 0; kk < HCH; kk += 32) {
    const size_t fo = (size_t)(kk >> 5) * (8 * 32 * 8) + lbase;
    BF16x16 aW, aU, aC;
    aW.q[0] = *(const uint4*)(wA0 + fo); aW.q[1] = *(const uint4*)(wA0 + fo + 4);
    aU.q[0] = *(const uint4*)(wA1 + fo); aU.q[1] = *(const uint4*)(wA1 + fo + 4);
    aC.q[0] = *(const uint4*)(wA2 + fo); aC.q[1] = *(const uint4*)(wA2 + fo + 4);
#pragma unroll
    for (int tile = 0; tile < 2; ++tile) {
      BF16x16 bB0, bB1;
#pragma unroll
      for (int j = 0; j < 8; ++j) {          // B 32x16 bf16 layout
        int kidx = kk + frag_k(hi, j);
        bB0.u[j] = *(const unsigned int*)(&sB0[(row + 16 * tile) * HCH + kidx]);
        bB1.u[j] = *(const unsigned int*)(&sB1[(row + 16 * tile) * HCH + kidx]);
      }
      accW[tile] = __builtin_amdgcn_wmma_f32_16x16x32_bf16(false, aW.v, false, bB0.v, (short)0, accW[tile], false, false);
      accU[tile] = __builtin_amdgcn_wmma_f32_16x16x32_bf16(false, aU.v, false, bB1.v, (short)0, accU[tile], false, false);
      accC[tile] = __builtin_amdgcn_wmma_f32_16x16x32_bf16(false, aC.v, false, bB1.v, (short)0, accC[tile], false, false);
    }
  }

  const float stepf = (float)(*pstep);
#pragma unroll
  for (int tile = 0; tile < 2; ++tile) {
    const int col = row + 16 * tile;
#pragma unroll
    for (int r = 0; r < 8; ++r) {
      int m = m0 + r + 8 * hi;               // C/D layout: VGPR r -> M = r + 8*hi
      int p = p0 + col;
      size_t gidx = (size_t)b * HCH * HW + (size_t)m * HW + (hw0 + col);
      float gw = accW[tile][r] + bias0[m];
      float gu = accU[tile][r] + bias1[m];
      float gc = accC[tile][r] + bias2[m];
      float cs = __cosf(gc * stepf);
      float gatev = cs * cs * sigmoidf_(gw + gu);
      if (MODE == 0) {
        float e = srcB1[gidx];
        outBF[(size_t)m * NPIX + p] = f2bf(e * gatev);
      } else if (MODE == 1) {
        float mean = stats[m], inv = stats[HCH + m];
        float intx = (convraw[(size_t)m * NPIX + p] - mean) * inv * bnw[m] + bnb[m];
        float inh = srcB1[gidx];
        float x   = srcB0[gidx];
        float ih  = tanhf(x - tanhf(intx * (c0[m] * inh + c1[m])));   // alpha, mu
        float nv  = (1.f - gatev) * inh + gatev * ih;
        outF[gidx] = nv;
        outBF[(size_t)m * NPIX + p] = f2bf(nv);
      } else {
        float mean = stats[m], inv = stats[HCH + m];
        float intx = (convraw[(size_t)m * NPIX + p] - mean) * inv * bnw[m] + bnb[m];
        float inew = srcB0[gidx];
        float eold = srcB1[gidx];
        float eh   = tanhf(c0[m] * inew + c1[m] * intx + c2[m] * inew * intx); // kappa,gamma,w_gain
        float nv   = (1.f - gatev) * eold + gatev * eh;
        outF[gidx] = nv;
      }
    }
  }
}

// ---------------------------------------------------------------------------
// 5x5 'same' conv 128->128 as implicit GEMM (K = 128*25), + BN partial stats.
// Two pixel-tiles per wave: each A fragment (1 KB/wave) feeds two WMMAs.
// Block: 32 pixels (row segment) x 128 out-channels (8 waves).
// ---------------------------------------------------------------------------
__global__ __launch_bounds__(256)
void conv5_kernel(const unsigned short* __restrict__ inBF,   // [c][NPIX] bf16
                  const unsigned int* __restrict__ wPK,      // packed fragments
                  float* __restrict__ outF,                  // [o][NPIX] raw conv
                  float* __restrict__ gstats) {              // [2*HCH] sum,sumsq
  __shared__ unsigned short sIn[5 * 36 * 32];   // [dy][xx][ci] ci-pairs contiguous
  __shared__ float sstat[2 * HCH];

  const int tid = threadIdx.x;
  const int p0  = blockIdx.x * 32;
  const int b   = p0 / HW;
  const int hw0 = p0 - b * HW;
  const int y   = hw0 / SS;
  const int x0  = hw0 - y * SS;                  // 0, 32, 64
  sstat[tid] = 0.f;                              // 256 == 2*HCH

  const int wave = tid >> 5, lane = tid & 31;
  const int row = lane & 15, hi = lane >> 4;
  const int m0 = wave * 16;
  const size_t lbase = ((size_t)wave * 32 + lane) * 8;

  v8f acc[2] = {{}, {}};
  for (int ic0 = 0; ic0 < HCH; ic0 += 32) {
    __syncthreads();
    for (int idx = tid; idx < 5 * 36 * 32; idx += 256) {
      int ci = idx & 31;
      int rr = idx >> 5;                         // 0..179
      int xx = rr % 36;
      int dy = rr / 36;
      int yy = y + dy - PAD;
      int xg = x0 + xx - PAD;
      unsigned short v = 0;
      if (yy >= 0 && yy < SS && xg >= 0 && xg < SS)
        v = inBF[(size_t)(ic0 + ci) * NPIX + (size_t)b * HW + yy * SS + xg];
      sIn[(dy * 36 + xx) * 32 + ci] = v;
    }
    if (ic0 + 32 < HCH)                          // hint next chunk into cache
      __builtin_prefetch(inBF + (size_t)(ic0 + 32 + (tid & 31)) * NPIX +
                         (size_t)b * HW + hw0, 0, 0);
    __syncthreads();

#pragma unroll
    for (int dy = 0; dy < 5; ++dy) {
#pragma unroll
      for (int dx = 0; dx < 5; ++dx) {
        const int t = dy * 5 + dx;
        BF16x16 A;
        const unsigned int* pw =
            wPK + (size_t)(t * 4 + (ic0 >> 5)) * (8 * 32 * 8) + lbase;
        A.q[0] = *(const uint4*)pw;
        A.q[1] = *(const uint4*)(pw + 4);
#pragma unroll
        for (int tile = 0; tile < 2; ++tile) {
          BF16x16 Bm;
#pragma unroll
          for (int j = 0; j < 8; ++j) {
            int kidx = frag_k(hi, j);
            Bm.u[j] = *(const unsigned int*)(
                &sIn[(dy * 36 + (row + dx + 16 * tile)) * 32 + kidx]);
          }
          acc[tile] = __builtin_amdgcn_wmma_f32_16x16x32_bf16(false, A.v, false, Bm.v, (short)0, acc[tile], false, false);
        }
      }
    }
  }

#pragma unroll
  for (int r = 0; r < 8; ++r) {
    int m = m0 + r + 8 * hi;
    float v0 = acc[0][r];
    float v1 = acc[1][r];
    outF[(size_t)m * NPIX + (p0 + row)]      = v0;
    outF[(size_t)m * NPIX + (p0 + 16 + row)] = v1;
    atomicAdd(&sstat[m], v0 + v1);
    atomicAdd(&sstat[HCH + m], v0 * v0 + v1 * v1);
  }
  __syncthreads();
  atomicAdd(&gstats[tid], sstat[tid]);
}

// ---------------------------------------------------------------------------
// Convert {sum, sumsq} -> {mean, invstd} (biased var, matches jnp.var)
// ---------------------------------------------------------------------------
__global__ void bnstats_kernel(float* __restrict__ stats) {
  int c = threadIdx.x;
  if (c < HCH) {
    float s = stats[c], ss = stats[HCH + c];
    float mean = s / (float)NPIX;
    float var = ss / (float)NPIX - mean * mean;
    stats[c] = mean;
    stats[HCH + c] = rsqrtf(var + EPSB);
  }
}

// ---------------------------------------------------------------------------
extern "C" void kernel_launch(void* const* d_in, const int* in_sizes, int n_in,
                              void* d_out, int out_size, void* d_ws, size_t ws_size,
                              hipStream_t stream) {
  const float* x    = (const float*)d_in[0];
  const float* inh  = (const float*)d_in[1];
  const float* exc  = (const float*)d_in[2];
  const float* aw_w = (const float*)d_in[3];  const float* aw_b = (const float*)d_in[4];
  const float* au_w = (const float*)d_in[5];  const float* au_b = (const float*)d_in[6];
  const float* iw_w = (const float*)d_in[7];  const float* iw_b = (const float*)d_in[8];
  const float* iu_w = (const float*)d_in[9];  const float* iu_b = (const float*)d_in[10];
  const float* ew_w = (const float*)d_in[11]; const float* ew_b = (const float*)d_in[12];
  const float* eu_w = (const float*)d_in[13]; const float* eu_b = (const float*)d_in[14];
  const float* ac_w = (const float*)d_in[15]; const float* ac_b = (const float*)d_in[16];
  const float* ic_w = (const float*)d_in[17]; const float* ic_b = (const float*)d_in[18];
  const float* ec_w = (const float*)d_in[19]; const float* ec_b = (const float*)d_in[20];
  const float* w_inh = (const float*)d_in[21];
  const float* w_exc = (const float*)d_in[22];
  const float* alpha = (const float*)d_in[23];
  const float* gamma_ = (const float*)d_in[24];
  const float* kappa = (const float*)d_in[25];
  const float* w_gain = (const float*)d_in[26];
  const float* mu = (const float*)d_in[27];
  const float* bn0_w = (const float*)d_in[28]; const float* bn0_b = (const float*)d_in[29];
  const float* bn1_w = (const float*)d_in[30]; const float* bn1_b = (const float*)d_in[31];
  const int*   pstep = (const int*)d_in[32];

  float* out_inh = (float*)d_out;
  float* out_exc = (float*)d_out + NTOT;

  // workspace layout
  char* ws = (char*)d_ws;
  size_t off = 0;
  float* conv_out = (float*)(ws + off);               off += NTOT * 4;   // raw conv
  unsigned short* gbuf = (unsigned short*)(ws + off); off += NTOT * 2;   // bf16 activations
  float* stats0 = (float*)(ws + off);                 off += 2 * HCH * 4;
  float* stats1 = (float*)(ws + off);                 off += 2 * HCH * 4;
  const size_t W1P = 8192;          // u32 per packed 1x1 matrix
  const size_t WKP = 25 * 8192;     // u32 per packed 5x5 kernel
  unsigned int* wp = (unsigned int*)(ws + off);
  unsigned int* p_aw = wp;            unsigned int* p_au = p_aw + W1P;
  unsigned int* p_ac = p_au + W1P;    unsigned int* p_iw = p_ac + W1P;
  unsigned int* p_iu = p_iw + W1P;    unsigned int* p_ic = p_iu + W1P;
  unsigned int* p_ew = p_ic + W1P;    unsigned int* p_eu = p_ew + W1P;
  unsigned int* p_ec = p_eu + W1P;
  unsigned int* p_wi = p_ec + W1P;    unsigned int* p_we = p_wi + WKP;

  hipMemsetAsync(stats0, 0, 2 * 2 * HCH * sizeof(float), stream);  // stats0+stats1

  auto pack1 = [&](const float* s, unsigned int* d) {
    pack_w1_kernel<<<8192 / 256, 256, 0, stream>>>(s, d);
  };
  pack1(aw_w, p_aw); pack1(au_w, p_au); pack1(ac_w, p_ac);
  pack1(iw_w, p_iw); pack1(iu_w, p_iu); pack1(ic_w, p_ic);
  pack1(ew_w, p_ew); pack1(eu_w, p_eu); pack1(ec_w, p_ec);
  pack_wk_kernel<<<(25 * 8192) / 256, 256, 0, stream>>>(w_inh, p_wi);
  pack_wk_kernel<<<(25 * 8192) / 256, 256, 0, stream>>>(w_exc, p_we);

  const int NT = NPIX / 32;   // 2304 tiles of 32 pixels
  dim3 grid(NT), blk(256);

  // 1) attention gate: gbuf = excitation * att_gate (bf16)
  gate_kernel<0><<<grid, blk, 0, stream>>>(
      x, exc, p_aw, p_au, p_ac, aw_b, au_b, ac_b, pstep,
      nullptr, nullptr, nullptr, nullptr, nullptr, nullptr, nullptr,
      nullptr, gbuf);

  // 2) conv5x5(gbuf, w_inh) + BN0 stats
  conv5_kernel<<<grid, blk, 0, stream>>>(gbuf, p_wi, conv_out, stats0);
  bnstats_kernel<<<1, HCH, 0, stream>>>(stats0);

  // 3) inhibition update -> out_inh (f32) + gbuf (bf16)
  gate_kernel<1><<<grid, blk, 0, stream>>>(
      x, inh, p_iw, p_iu, p_ic, iw_b, iu_b, ic_b, pstep,
      conv_out, stats0, bn0_w, bn0_b, alpha, mu, nullptr,
      out_inh, gbuf);

  // 4) conv5x5(inh_new, w_exc) + BN1 stats
  conv5_kernel<<<grid, blk, 0, stream>>>(gbuf, p_we, conv_out, stats1);
  bnstats_kernel<<<1, HCH, 0, stream>>>(stats1);

  // 5) excitation update -> out_exc (f32)
  gate_kernel<2><<<grid, blk, 0, stream>>>(
      out_inh, exc, p_ew, p_eu, p_ec, ew_b, eu_b, ec_b, pstep,
      conv_out, stats1, bn1_w, bn1_b, kappa, gamma_, w_gain,
      out_exc, nullptr);

  (void)in_sizes; (void)n_in; (void)out_size; (void)ws_size;
}
